// LstmEncDeltaStackedFullPredMultyGaus_48670569398994
// MI455X (gfx1250) — matrix operand order repaired and checked by reading, not compiled, not run on hardware
//
#include <hip/hip_runtime.h>

// ---------- CDNA5 WMMA plumbing (gfx1250, wave32) ----------
typedef __attribute__((ext_vector_type(16))) __bf16        v16bf;
typedef __attribute__((ext_vector_type(8)))  float         v8f;
typedef __attribute__((ext_vector_type(8)))  unsigned int  v8u;

__device__ __forceinline__ unsigned short f2bf(float f) {
  unsigned u = __float_as_uint(f);
  return (unsigned short)((u + 0x7FFFu + ((u >> 16) & 1u)) >> 16); // RNE
}
__device__ __forceinline__ unsigned pk2bf(float a, float b) {
  return (unsigned)f2bf(a) | ((unsigned)f2bf(b) << 16);
}
__device__ __forceinline__ v8f wmma_bf16(v8u a, v8u b, v8f c) {
  return __builtin_amdgcn_wmma_f32_16x16x32_bf16(
      false, __builtin_bit_cast(v16bf, a),
      false, __builtin_bit_cast(v16bf, b),
      (short)0, c, false, false);
}
// fast rcp (v_rcp_f32) -- sigmoid/tanh tolerate ~1ulp rcp; keeps the
// latency-critical recurrent chain short (no div_scale/newton sequence).
__device__ __forceinline__ float frcp(float x) { return __builtin_amdgcn_rcpf(x); }
__device__ __forceinline__ float sigm(float x) { return frcp(1.f + __expf(-x)); }
__device__ __forceinline__ float tanh_(float x) {
  float e = __expf(-2.f * fabsf(x));
  float t = (1.f - e) * frcp(1.f + e);
  return copysignf(t, x);
}
__device__ __forceinline__ unsigned mix32(unsigned x) {
  x ^= x >> 16; x *= 0x7feb352du; x ^= x >> 15; x *= 0x846ca68bu; x ^= x >> 16; return x;
}
__device__ __forceinline__ float u01(unsigned r) {
  return ((float)(r >> 8) + 0.5f) * (1.f / 16777216.f);
}

// ---------- P[t,col] = cur[t] @ Wih_edge_f.T  (512 x 256) ----------
__global__ void edge_p_precompute(const float* __restrict__ scene,
                                  const float* __restrict__ eWih,
                                  float* __restrict__ P) {
  int idx = blockIdx.x * 256 + threadIdx.x;   // 512 blocks x 256 thr
  int t = idx >> 8, col = idx & 255;
  float c0 = scene[(t * 8 + 7) * 4 + 0];
  float c1 = scene[(t * 8 + 7) * 4 + 1];
  P[idx] = c0 * eWih[col * 2] + c1 * eWih[col * 2 + 1];
}

// ---------- Forward LSTM via WMMA (node: 8 steps, edge: 512 steps) ----------
// One block = 16 batch rows, 4 waves. Wave w owns gate-column tiles
// {w, 4+w, 8+w, 12+w} == gates i/f/g/o for hidden j in [16w, 16w+16).
template <int STEPS, bool EDGE>
__global__ __launch_bounds__(128) void lstm_fwd_wmma(
    const float* __restrict__ scene, const float* __restrict__ Wih,
    const float* __restrict__ Whh, const float* __restrict__ bih,
    const float* __restrict__ bhh, const float* __restrict__ P,
    float* __restrict__ catted, int outBase) {
  __shared__ unsigned short hbuf[2][16 * 64];   // bf16 h, double buffered
  __shared__ float smemF[16 * 256];             // EDGE: bias - P[row]; node: x stage

  const int tid = threadIdx.x;
  const int L = tid & 31, w = tid >> 5;
  const int lm = L & 15, half = L >> 4;
  const int rowbase = blockIdx.x * 16;

  for (int i = tid; i < 16 * 64; i += 128) hbuf[0][i] = 0;
  if (EDGE) {
    for (int i = tid; i < 16 * 256; i += 128) {
      int m = i >> 8, col = i & 255;
      smemF[i] = bih[col] + bhh[col] - P[(rowbase + m) * 256 + col];
    }
  } else {
    for (int i = tid; i < 16 * 8 * 2; i += 128) {  // xs[m][t][c]
      int m = i >> 4, t = (i >> 1) & 7, c = i & 1;
      smemF[i] = scene[((rowbase + m) * 8 + t) * 4 + 2 + c];
    }
  }

  // B fragments: WhhT tiles (k,n) with B[k,n] = Whh[n,k], bf16, once.
  v8u Bf[4][2];
  float wih0[4], wih1[4], bsum[4];
#pragma unroll
  for (int fi = 0; fi < 4; ++fi) {
    int col = (fi * 4 + w) * 16 + lm;  // gate column 0..255
    if (!EDGE) {
      wih0[fi] = Wih[col * 2];
      wih1[fi] = Wih[col * 2 + 1];
      bsum[fi] = bih[col] + bhh[col];
    }
#pragma unroll
    for (int kt = 0; kt < 2; ++kt) {
      const float* wr = Whh + col * 64 + kt * 32 + half * 16;  // k = kt*32+half*16+2v+p
#pragma unroll
      for (int v = 0; v < 8; ++v) Bf[fi][kt][v] = pk2bf(wr[2 * v], wr[2 * v + 1]);
    }
  }
  __syncthreads();

  v8f cst, hlast;
#pragma unroll
  for (int v = 0; v < 8; ++v) { cst[v] = 0.f; hlast[v] = 0.f; }

  const int j = w * 16 + lm;  // this wave's hidden-unit index
  int curb = 0;
  for (int t = 0; t < STEPS; ++t) {
    // C init = bias + x@WihT  (EDGE: P[t,col] + (bias - P[row,col]))
    v8f acc[4];
#pragma unroll
    for (int fi = 0; fi < 4; ++fi) {
      int col = (fi * 4 + w) * 16 + lm;
      if (EDGE) {
        float pv = P[t * 256 + col];
#pragma unroll
        for (int v = 0; v < 8; ++v) acc[fi][v] = pv + smemF[(v + half * 8) * 256 + col];
      } else {
#pragma unroll
        for (int v = 0; v < 8; ++v) {
          int m = v + half * 8;
          acc[fi][v] = bsum[fi] + smemF[m * 16 + t * 2] * wih0[fi] +
                       smemF[m * 16 + t * 2 + 1] * wih1[fi];
        }
      }
    }
    if (EDGE && t + 1 < STEPS)
      __builtin_prefetch(&P[(t + 1) * 256 + (tid << 1)], 0, 1);  // global_prefetch_b8

    // A fragments from bf16 h in LDS (16-bit A 16x32 layout, 2x b128 per ktile)
    const unsigned short* hb = hbuf[curb];
    int abase = lm * 64;
    uint4 a0 = *(const uint4*)(hb + abase + half * 16);
    uint4 a1 = *(const uint4*)(hb + abase + 8 + half * 16);
    uint4 a2 = *(const uint4*)(hb + abase + 32 + half * 16);
    uint4 a3 = *(const uint4*)(hb + abase + 40 + half * 16);
    v8u A0, A1;
    A0[0] = a0.x; A0[1] = a0.y; A0[2] = a0.z; A0[3] = a0.w;
    A0[4] = a1.x; A0[5] = a1.y; A0[6] = a1.z; A0[7] = a1.w;
    A1[0] = a2.x; A1[1] = a2.y; A1[2] = a2.z; A1[3] = a2.w;
    A1[4] = a3.x; A1[5] = a3.y; A1[6] = a3.z; A1[7] = a3.w;

#pragma unroll
    for (int fi = 0; fi < 4; ++fi) {
      acc[fi] = wmma_bf16(A0, Bf[fi][0], acc[fi]);
      acc[fi] = wmma_bf16(A1, Bf[fi][1], acc[fi]);
    }

    // gate nonlinearities + c/h update (in-register; acc[0..3] = i,f,g,o)
    int nxt = curb ^ 1;
#pragma unroll
    for (int v = 0; v < 8; ++v) {
      float si = sigm(acc[0][v]);
      float sf = sigm(acc[1][v]);
      float tg = tanh_(acc[2][v]);
      float so = sigm(acc[3][v]);
      float c2 = sf * cst[v] + si * tg;
      cst[v] = c2;
      float h2 = so * tanh_(c2);
      hlast[v] = h2;
      hbuf[nxt][(v + half * 8) * 64 + j] = f2bf(h2);
    }
    __syncthreads();
    curb = nxt;
  }
#pragma unroll
  for (int v = 0; v < 8; ++v)
    catted[(rowbase + v + half * 8) * 256 + outBase + j] = hlast[v];
}

// ---------- Backward-direction LSTMs collapse to one step from h=c=0 ----------
__global__ __launch_bounds__(128) void lstm_bwd_single(
    const float* __restrict__ scene, const float* __restrict__ nWih,
    const float* __restrict__ nbih, const float* __restrict__ nbhh,
    const float* __restrict__ eWih, const float* __restrict__ ebih,
    const float* __restrict__ ebhh, float* __restrict__ catted) {
  int b = blockIdx.x, tid = threadIdx.x;
  int jj = tid & 63;
  bool edge = tid >= 64;
  float x0, x1;
  const float *W, *bi, *bh;
  int ob;
  if (edge) {
    x0 = scene[(511 * 8 + 7) * 4 + 0] - scene[(b * 8 + 7) * 4 + 0];
    x1 = scene[(511 * 8 + 7) * 4 + 1] - scene[(b * 8 + 7) * 4 + 1];
    W = eWih; bi = ebih; bh = ebhh; ob = 192;
  } else {
    x0 = scene[(b * 8 + 7) * 4 + 2];
    x1 = scene[(b * 8 + 7) * 4 + 3];
    W = nWih; bi = nbih; bh = nbhh; ob = 64;
  }
  float ig = bi[jj] + bh[jj] + x0 * W[2 * jj] + x1 * W[2 * jj + 1];
  int k = 128 + jj;
  float gg = bi[k] + bh[k] + x0 * W[2 * k] + x1 * W[2 * k + 1];
  k = 192 + jj;
  float og = bi[k] + bh[k] + x0 * W[2 * k] + x1 * W[2 * k + 1];
  float c2 = sigm(ig) * tanh_(gg);  // f-gate * 0 drops out
  catted[b * 256 + ob + jj] = sigm(og) * tanh_(c2);
}

// ---------- Fused GRU decoder: 12 iterations, one block per batch row ----------
__global__ __launch_bounds__(128) void decoder_kernel(
    const float* __restrict__ scene, const float* __restrict__ catted,
    const float* __restrict__ gWih, const float* __restrict__ gWhh,
    const float* __restrict__ gbih, const float* __restrict__ gbhh,
    const float* __restrict__ aW, const float* __restrict__ ab,
    const float* __restrict__ sW, const float* __restrict__ sb,
    const float* __restrict__ pW, const float* __restrict__ pb,
    const float* __restrict__ mW, const float* __restrict__ mb,
    const float* __restrict__ sgW, const float* __restrict__ sgb,
    const float* __restrict__ cW, const float* __restrict__ cb,
    float* __restrict__ out) {
  __shared__ float inp[258];
  __shared__ float st[40], gi[120], gh[120], hh[40], cs[40];
  __shared__ float mus[40], lsig[40], varr[40], lp[20], corrs[20];
  __shared__ float sA[2];
  __shared__ float sLse;

  int b = blockIdx.x, tid = threadIdx.x;
  for (int i = tid; i < 256; i += 128) inp[i] = catted[b * 256 + i];
  float cur0 = scene[(b * 8 + 7) * 4 + 0];
  float cur1 = scene[(b * 8 + 7) * 4 + 1];
  if (tid == 0) {
    sA[0] = aW[0] * cur0 + aW[1] * cur1 + ab[0];
    sA[1] = aW[2] * cur0 + aW[3] * cur1 + ab[1];
  }
  if (tid < 40) cs[tid] = (tid & 1) ? cur1 : cur0;
  __syncthreads();
  if (tid < 40) {  // state0 = catted @ state_W.T + state_b
    float s = sb[tid];
    const float* wr = sW + tid * 256;
    for (int k = 0; k < 256; ++k) s += inp[k] * wr[k];
    st[tid] = s;
  }
  __syncthreads();

  for (int it = 0; it < 12; ++it) {
    if (tid < 2) inp[256 + tid] = sA[tid];
    __syncthreads();
    if (tid < 120) {
      float s = gbih[tid];
      const float* wr = gWih + tid * 258;
      for (int k = 0; k < 258; ++k) s += inp[k] * wr[k];
      gi[tid] = s;
      float s2 = gbhh[tid];
      const float* wr2 = gWhh + tid * 40;
      for (int k = 0; k < 40; ++k) s2 += st[k] * wr2[k];
      gh[tid] = s2;
    }
    __syncthreads();
    if (tid < 40) {
      float r = sigm(gi[tid] + gh[tid]);
      float z = sigm(gi[40 + tid] + gh[40 + tid]);
      float n = tanh_(gi[80 + tid] + r * gh[80 + tid]);
      hh[tid] = (1.f - z) * n + z * st[tid];
    }
    __syncthreads();
    if (tid < 120) {  // heads, dot-40 over h
      if (tid < 20) {
        float s = pb[tid];
        const float* wr = pW + tid * 40;
        for (int k = 0; k < 40; ++k) s += hh[k] * wr[k];
        lp[tid] = s;
      } else if (tid < 60) {
        int j2 = tid - 20;
        float s = mb[j2];
        const float* wr = mW + j2 * 40;
        for (int k = 0; k < 40; ++k) s += hh[k] * wr[k];
        s = fminf(fmaxf(s, -1.5f), 1.5f);
        mus[j2] = s + cs[j2];
      } else if (tid < 100) {
        int j2 = tid - 60;
        float s = sgb[j2];
        const float* wr = sgW + j2 * 40;
        for (int k = 0; k < 40; ++k) s += hh[k] * wr[k];
        lsig[j2] = s;
        float e = __expf(s);
        varr[j2] = fminf(e * e, 1000.f);
      } else {
        int j2 = tid - 100;
        float s = cb[j2];
        const float* wr = cW + j2 * 40;
        for (int k = 0; k < 40; ++k) s += hh[k] * wr[k];
        corrs[j2] = tanh_(s);
      }
    }
    __syncthreads();
    if (tid == 0) {  // log-softmax normalizer + Gumbel-max sample + normal noise
      float mx = -1e30f;
      for (int k = 0; k < 20; ++k) mx = fmaxf(mx, lp[k]);
      float se = 0.f;
      for (int k = 0; k < 20; ++k) se += __expf(lp[k] - mx);
      sLse = mx + __logf(se);
      unsigned seed = mix32(0x9E3779B9u ^ (unsigned)(b * 7919 + it * 104729));
      int comp = 0;
      float best = -1e30f;
      for (int k = 0; k < 20; ++k) {
        seed = mix32(seed + 0x85EBCA6Bu);
        float u = u01(seed);
        float g = -__logf(fmaxf(-__logf(u), 1e-20f));
        float val = (lp[k] - sLse) + g;
        if (val > best) { best = val; comp = k; }
      }
      seed = mix32(seed + 0xC2B2AE35u);
      float u1 = u01(seed);
      seed = mix32(seed + 0x27D4EB2Fu);
      float u2 = u01(seed);
      float rr = sqrtf(-2.f * __logf(fmaxf(u1, 1e-20f)));
      float e0 = rr * __cosf(6.2831853f * u2);
      float e1 = rr * __sinf(6.2831853f * u2);
      sA[0] = mus[comp * 2 + 0] + sqrtf(varr[comp * 2 + 0]) * e0;
      sA[1] = mus[comp * 2 + 1] + sqrtf(varr[comp * 2 + 1]) * e1;
    }
    __syncthreads();
    float* op = out + ((size_t)it * 512 + b) * 120;
    if (tid < 20) {
      op[tid] = lp[tid] - sLse;
      op[100 + tid] = corrs[tid];
    }
    if (tid < 40) {
      op[20 + tid] = mus[tid];
      op[60 + tid] = lsig[tid];
      st[tid] = hh[tid];
      cs[tid] = mus[tid];
    }
    __syncthreads();
  }
}

extern "C" void kernel_launch(void* const* d_in, const int* in_sizes, int n_in,
                              void* d_out, int out_size, void* d_ws, size_t ws_size,
                              hipStream_t stream) {
  (void)in_sizes; (void)n_in; (void)out_size; (void)ws_size;
  const float* scene = (const float*)d_in[0];
  const float* nWihf = (const float*)d_in[1];
  const float* nWhhf = (const float*)d_in[2];
  const float* nbihf = (const float*)d_in[3];
  const float* nbhhf = (const float*)d_in[4];
  const float* nWihb = (const float*)d_in[5];
  const float* nbihb = (const float*)d_in[7];
  const float* nbhhb = (const float*)d_in[8];
  const float* eWihf = (const float*)d_in[9];
  const float* eWhhf = (const float*)d_in[10];
  const float* ebihf = (const float*)d_in[11];
  const float* ebhhf = (const float*)d_in[12];
  const float* eWihb = (const float*)d_in[13];
  const float* ebihb = (const float*)d_in[15];
  const float* ebhhb = (const float*)d_in[16];
  const float* gWih = (const float*)d_in[17];
  const float* gWhh = (const float*)d_in[18];
  const float* gbih = (const float*)d_in[19];
  const float* gbhh = (const float*)d_in[20];
  const float* aW = (const float*)d_in[21];
  const float* ab = (const float*)d_in[22];
  const float* sW = (const float*)d_in[23];
  const float* sb = (const float*)d_in[24];
  const float* pW = (const float*)d_in[25];
  const float* pb = (const float*)d_in[26];
  const float* mW = (const float*)d_in[27];
  const float* mb = (const float*)d_in[28];
  const float* sgW = (const float*)d_in[29];
  const float* sgb = (const float*)d_in[30];
  const float* cW = (const float*)d_in[31];
  const float* cb = (const float*)d_in[32];

  float* P = (float*)d_ws;          // 512*256 f32
  float* catted = P + 512 * 256;    // 512*256 f32

  edge_p_precompute<<<512, 256, 0, stream>>>(scene, eWihf, P);
  lstm_fwd_wmma<8, false><<<32, 128, 0, stream>>>(scene, nWihf, nWhhf, nbihf, nbhhf, P, catted, 0);
  lstm_fwd_wmma<512, true><<<32, 128, 0, stream>>>(scene, eWihf, eWhhf, ebihf, ebhhf, P, catted, 128);
  lstm_bwd_single<<<512, 128, 0, stream>>>(scene, nWihb, nbihb, nbhhb, eWihb, ebihb, ebhhb, catted);
  decoder_kernel<<<512, 128, 0, stream>>>(scene, catted, gWih, gWhh, gbih, gbhh, aW, ab,
                                          sW, sb, pW, pb, mW, mb, sgW, sgb, cW, cb,
                                          (float*)d_out);
}
